// CharnerRNNModel_27023934226950
// MI455X (gfx1250) — compile-verified, matrix operand order checked by reading, not run on hardware
//
#include <hip/hip_runtime.h>

typedef __attribute__((ext_vector_type(16))) _Float16 v16h;
typedef __attribute__((ext_vector_type(8)))  float    v8f;
typedef __attribute__((ext_vector_type(4)))  int      v4i;

constexpr int BB   = 32;     // batch
constexpr int TT   = 1024;   // time
constexpr int EMBD = 128;
constexpr int HID  = 256;
constexpr int G4   = 1024;   // 4*HID (gates)
constexpr int NOUT = 5;
// LDS: h(16K) + c(32K) + XW/z double buffer(2x128K) + lengths + anti-hoist ptr
constexpr size_t SMEM_BYTES = 304 * 1024 + 256;

#if defined(__has_builtin)
#if __has_builtin(__builtin_amdgcn_global_load_async_to_lds_b128)
#define HAVE_ASYNC_LDS 1
#endif
#endif
#ifndef HAVE_ASYNC_LDS
#define HAVE_ASYNC_LDS 0
#warning "gfx1250 async-to-LDS builtin not found; using synchronous LDS staging fallback"
#endif

#if HAVE_ASYNC_LDS
typedef __attribute__((address_space(1))) v4i gv4i_t;   // global
typedef __attribute__((address_space(3))) v4i lv4i_t;   // LDS
#endif

static __device__ __forceinline__ float sigmoidf_(float x) {
  return 1.0f / (1.0f + __expf(-x));
}

// copy 16 bytes global -> LDS (async on the ASYNCcnt path when available)
static __device__ __forceinline__ void copy16_g2l(const float* g, float* l) {
#if HAVE_ASYNC_LDS
  __builtin_amdgcn_global_load_async_to_lds_b128(
      (gv4i_t*)(float*)g, (lv4i_t*)l, 0, 0);
#else
  *(float4*)l = *(const float4*)g;
#endif
}

static __device__ __forceinline__ void wait_async0() {
#if HAVE_ASYNC_LDS
#if __has_builtin(__builtin_amdgcn_s_wait_asynccnt)
  __builtin_amdgcn_s_wait_asynccnt(0);
#else
  asm volatile("s_wait_asynccnt 0x0" ::: "memory");
#endif
#endif
}

// ---------------- embedding lookup -> f16 ----------------
__global__ void embed_kernel(const int* __restrict__ tokens,
                             const float* __restrict__ emb,
                             _Float16* __restrict__ x0) {
  int i = blockIdx.x * blockDim.x + threadIdx.x;
  if (i < BB * TT * EMBD) {
    int row = i >> 7;
    int e   = i & 127;
    x0[i] = (_Float16)emb[tokens[row] * EMBD + e];
  }
}

// ------- transpose + f16-convert weights: W(in+H,4H) -> WxT(4H,in), WhT(4H,H) -------
__global__ void prep_weights_kernel(const float* __restrict__ fw_W, const float* __restrict__ fw_b,
                                    const float* __restrict__ bw_W, const float* __restrict__ bw_b,
                                    _Float16* __restrict__ WxT, _Float16* __restrict__ WhT,
                                    float* __restrict__ biasF, int in_dim) {
  int stride = gridDim.x * blockDim.x;
  int rows = in_dim + HID;
  long total = 2L * G4 * rows;
  for (long idx = blockIdx.x * (long)blockDim.x + threadIdx.x; idx < total; idx += stride) {
    int  dir = (int)(idx / ((long)G4 * rows));
    long rem = idx % ((long)G4 * rows);
    int  n = (int)(rem / rows);
    int  k = (int)(rem % rows);
    const float* W = dir ? bw_W : fw_W;
    float v = W[(long)k * G4 + n];
    if (k < in_dim) WxT[(long)dir * G4 * in_dim + (long)n * in_dim + k] = (_Float16)v;
    else            WhT[(long)dir * G4 * HID + (long)n * HID + (k - in_dim)] = (_Float16)v;
  }
  for (int idx = blockIdx.x * blockDim.x + threadIdx.x; idx < 2 * G4; idx += stride) {
    const float* bsrc = (idx < G4) ? fw_b : bw_b;
    biasF[idx] = bsrc[idx & (G4 - 1)];
  }
}

// ------- XW = x @ Wx^T + bias  (M=B*T, N=4H, per-direction), f16 WMMA, f32 accum -------
// 256 threads = 8 waves; wave w owns a 16x128 strip; 8 v8f accumulators per wave.
// K compile-time: pipelined K-loop fully unrolls; generation copies are SSA renames.
template <int K>
__global__ __launch_bounds__(256) void xw_gemm_kernel(
    const _Float16* __restrict__ x,    // (B*T, K) row-major f16
    const _Float16* __restrict__ WxT,  // (2, G4, K) f16
    const float*    __restrict__ biasF,// (2, G4)
    float*          __restrict__ XW) { // (2, B*T, G4) f32
  const int lane  = threadIdx.x & 31;
  const int wave  = threadIdx.x >> 5;
  const int dir   = blockIdx.z;
  const int m0    = blockIdx.x * 128 + wave * 16;
  const int n0    = blockIdx.y * 128;
  const int lhalf = lane >> 4;
  const int lmod  = lane & 15;

  const _Float16* Arow  = x + (long)(m0 + lmod) * K;
  const _Float16* Bbase = WxT + (long)dir * G4 * K;

  v8f acc[8];
#pragma unroll
  for (int nt = 0; nt < 8; ++nt) {
    float bv = biasF[dir * G4 + n0 + nt * 16 + lmod];
    v8f a;
#pragma unroll
    for (int r = 0; r < 8; ++r) a[r] = bv;
    acc[nt] = a;
  }

  constexpr int ksteps = K >> 5;

  // prologue: fragments for kc = 0
  v16h a_cur, b_cur[8];
  {
    const int kbase = lhalf * 16;
    a_cur = *(const v16h*)(Arow + kbase);
#pragma unroll
    for (int nt = 0; nt < 8; ++nt)
      b_cur[nt] = *(const v16h*)(Bbase + (long)(n0 + nt * 16 + lmod) * K + kbase);
  }

#pragma unroll
  for (int kc = 0; kc < ksteps; ++kc) {
    v16h a_nxt, b_nxt[8];
    const bool more = (kc + 1) < ksteps;
    if (more) {
      const int kb2 = (kc + 1) * 32 + lhalf * 16;
      a_nxt = *(const v16h*)(Arow + kb2);
#pragma unroll
      for (int nt = 0; nt < 8; ++nt)
        b_nxt[nt] = *(const v16h*)(Bbase + (long)(n0 + nt * 16 + lmod) * K + kb2);
    }
#pragma unroll
    for (int nt = 0; nt < 8; ++nt)
      acc[nt] = __builtin_amdgcn_wmma_f32_16x16x32_f16(false, a_cur, false, b_cur[nt],
                                                       (short)0, acc[nt], false, false);
    if (more) {
      a_cur = a_nxt;
#pragma unroll
      for (int nt = 0; nt < 8; ++nt) b_cur[nt] = b_nxt[nt];
    }
  }

  float* C = XW + (long)dir * (BB * TT) * G4;
#pragma unroll
  for (int nt = 0; nt < 8; ++nt) {
#pragma unroll
    for (int r = 0; r < 8; ++r) {
      int m = m0 + lhalf * 8 + r;
      C[(long)m * G4 + n0 + nt * 16 + lmod] = acc[nt][r];
    }
  }
}

// ------- persistent per-direction LSTM with async double-buffered XW staging -------
__global__ __launch_bounds__(1024) void lstm_layer_kernel(
    const float*    __restrict__ XW,     // (2, B*T, G4) f32, bias folded in
    const _Float16* __restrict__ WhT,    // (2, G4, HID) f16
    const int*      __restrict__ lengths,// (B)
    _Float16*       __restrict__ y) {    // (B*T, 512): fw -> [0,256), bw -> [256,512)
  extern __shared__ char smem[];
  _Float16* h_lds   = (_Float16*)smem;                 // 32 x 256 f16   (16K)
  float*    c_lds   = (float*)(smem + 16 * 1024);      // 32 x 256 f32   (32K)
  char*     bufbase = smem + 48 * 1024;                // 2x (32 x 1024 f32) (2x128K)
  int*      len_lds = (int*)(smem + 304 * 1024);       // 32 ints
  // anti-hoist slot: Wh base pointer reloaded from LDS each timestep so the
  // compiler cannot prove the Wh fragment loads loop-invariant (avoids the
  // hoist->VGPR-overflow->scratch-spill pattern; Wh stays L2-resident anyway)
  const _Float16** wptr_lds = (const _Float16**)(smem + 304 * 1024 + 128);

  const int tid   = threadIdx.x;
  const int lane  = tid & 31;
  const int wave  = tid >> 5;
  const int dir   = blockIdx.x;
  const int lhalf = lane >> 4;
  const int lmod  = lane & 15;
  const int n0    = wave * 32;

  for (int i = tid; i < BB * HID; i += 1024) { h_lds[i] = (_Float16)0.f; c_lds[i] = 0.f; }
  if (tid < BB) len_lds[tid] = lengths[tid];
  if (tid == 0) *wptr_lds = WhT + (long)dir * G4 * HID;
  __syncthreads();

  const float* XWd = XW + (long)dir * (BB * TT) * G4;

  // stage XW tile for timestep ts into dst: 8 x 16B chunks per thread
  auto stage = [&](int ts, float* dst) {
#pragma unroll
    for (int i = 0; i < 8; ++i) {
      int chunk = tid + i * 1024;          // 16B chunks of the 128KB tile
      int b     = chunk >> 8;              // 256 chunks per 4KB row
      int off   = (chunk & 255) << 2;      // float offset within row
      int len   = len_lds[b];
      int ti    = (dir && ts < len) ? (len - 1 - ts) : ts;
      copy16_g2l(XWd + ((long)b * TT + ti) * G4 + off, dst + b * G4 + off);
    }
  };

  // prologue: bring XW[0] into buffer 0
  stage(0, (float*)bufbase);
  wait_async0();
  __syncthreads();

  for (int t = 0; t < TT; ++t) {
    float* buf_cur = (float*)(bufbase + (size_t)(t & 1) * 131072);
    float* buf_nxt = (float*)(bufbase + (size_t)((t + 1) & 1) * 131072);

    // overlap: start fetching next timestep's XW tile (ASYNCcnt)
    if (t + 1 < TT) stage(t + 1, buf_nxt);

    // opaque Wh base (reloaded from LDS each iteration; see comment above)
    const _Float16* Wb = *wptr_lds;

    // ---- z = XW[t] + h @ Wh (WMMA, K=256 -> 8 chunks), in-place in buf_cur ----
    v8f acc[2][2];
#pragma unroll
    for (int mt = 0; mt < 2; ++mt)
#pragma unroll
      for (int nt = 0; nt < 2; ++nt)
#pragma unroll
        for (int r = 0; r < 8; ++r)
          acc[mt][nt][r] = buf_cur[(mt * 16 + lhalf * 8 + r) * G4 + n0 + nt * 16 + lmod];
#pragma unroll
    for (int k = 0; k < 8; ++k) {
      v16h a0 = *(const v16h*)(h_lds + (lmod) * HID + k * 32 + lhalf * 16);
      v16h a1 = *(const v16h*)(h_lds + (16 + lmod) * HID + k * 32 + lhalf * 16);
#pragma unroll
      for (int nt = 0; nt < 2; ++nt) {
        v16h b = *(const v16h*)(Wb + (long)(n0 + nt * 16 + lmod) * HID + k * 32 + lhalf * 16);
        acc[0][nt] = __builtin_amdgcn_wmma_f32_16x16x32_f16(false, a0, false, b,
                                                            (short)0, acc[0][nt], false, false);
        acc[1][nt] = __builtin_amdgcn_wmma_f32_16x16x32_f16(false, a1, false, b,
                                                            (short)0, acc[1][nt], false, false);
      }
    }
    // each thread writes back exactly the elements it read as C-init: no cross-thread hazard
#pragma unroll
    for (int mt = 0; mt < 2; ++mt)
#pragma unroll
      for (int nt = 0; nt < 2; ++nt)
#pragma unroll
        for (int r = 0; r < 8; ++r)
          buf_cur[(mt * 16 + lhalf * 8 + r) * G4 + n0 + nt * 16 + lmod] = acc[mt][nt][r];
    __syncthreads();

    // ---- LSTM cell: 32x256 elements, 8 per thread ----
#pragma unroll
    for (int r = 0; r < 8; ++r) {
      int flat = tid + r * 1024;
      int b  = flat >> 8;
      int hc = flat & 255;
      float ig = buf_cur[b * G4 + hc];
      float jg = buf_cur[b * G4 + 256 + hc];
      float fg = buf_cur[b * G4 + 512 + hc];
      float og = buf_cur[b * G4 + 768 + hc];
      float cold = c_lds[b * HID + hc];
      float cnew = sigmoidf_(fg + 1.0f) * cold + sigmoidf_(ig) * tanhf(jg);
      float hnew = sigmoidf_(og) * tanhf(cnew);
      int  len = len_lds[b];
      bool msk = t < len;
      int  ti  = (dir && msk) ? (len - 1 - t) : t;
      if (msk) { c_lds[b * HID + hc] = cnew; h_lds[b * HID + hc] = (_Float16)hnew; }
      y[((long)b * TT + ti) * (2 * HID) + dir * HID + hc] =
          msk ? (_Float16)hnew : (_Float16)0.f;
    }
    wait_async0();   // drain this wave's async loads into buf_nxt
    __syncthreads();
  }
}

// ---------------- logits = concat @ out_W + out_b (N=5, plain VALU) ----------------
__global__ void out_proj_kernel(const _Float16* __restrict__ y2,
                                const float* __restrict__ out_W,
                                const float* __restrict__ out_b,
                                float* __restrict__ out) {
  int i = blockIdx.x * blockDim.x + threadIdx.x;
  if (i >= BB * TT * NOUT) return;
  int row = i / NOUT, j = i % NOUT;
  float acc = out_b[j];
  const _Float16* yr = y2 + (long)row * (2 * HID);
#pragma unroll
  for (int c = 0; c < (2 * HID) / 16; ++c) {
    v16h v = *(const v16h*)(yr + c * 16);
#pragma unroll
    for (int e = 0; e < 16; ++e)
      acc += (float)v[e] * out_W[(c * 16 + e) * NOUT + j];
  }
  out[i] = acc;
}

extern "C" void kernel_launch(void* const* d_in, const int* in_sizes, int n_in,
                              void* d_out, int out_size, void* d_ws, size_t ws_size,
                              hipStream_t stream) {
  const int*   tokens  = (const int*)d_in[0];
  const int*   lengths = (const int*)d_in[1];
  const float* emb     = (const float*)d_in[2];
  const float* fw_W0   = (const float*)d_in[3];
  const float* fw_b0   = (const float*)d_in[4];
  const float* bw_W0   = (const float*)d_in[5];
  const float* bw_b0   = (const float*)d_in[6];
  const float* fw_W1   = (const float*)d_in[7];
  const float* fw_b1   = (const float*)d_in[8];
  const float* bw_W1   = (const float*)d_in[9];
  const float* bw_b1   = (const float*)d_in[10];
  const float* out_W   = (const float*)d_in[11];
  const float* out_b   = (const float*)d_in[12];

  char* ws = (char*)d_ws;
  size_t off = 0;
  auto carve = [&](size_t bytes) -> char* {
    char* p = ws + off;
    off = (off + bytes + 255) & ~(size_t)255;
    return p;
  };
  _Float16* x0    = (_Float16*)carve((size_t)BB * TT * EMBD * 2);
  _Float16* y1    = (_Float16*)carve((size_t)BB * TT * 512 * 2);
  _Float16* y2    = (_Float16*)carve((size_t)BB * TT * 512 * 2);
  float*    XW    = (float*)   carve((size_t)2 * BB * TT * G4 * 4);
  _Float16* WxT0  = (_Float16*)carve((size_t)2 * G4 * EMBD * 2);
  _Float16* WhT0  = (_Float16*)carve((size_t)2 * G4 * HID * 2);
  float*    bias0 = (float*)   carve((size_t)2 * G4 * 4);
  _Float16* WxT1  = (_Float16*)carve((size_t)2 * G4 * 512 * 2);
  _Float16* WhT1  = (_Float16*)carve((size_t)2 * G4 * HID * 2);
  float*    bias1 = (float*)   carve((size_t)2 * G4 * 4);

  (void)hipFuncSetAttribute((const void*)lstm_layer_kernel,
                            hipFuncAttributeMaxDynamicSharedMemorySize, (int)SMEM_BYTES);

  embed_kernel<<<(BB * TT * EMBD + 255) / 256, 256, 0, stream>>>(tokens, emb, x0);
  prep_weights_kernel<<<512, 256, 0, stream>>>(fw_W0, fw_b0, bw_W0, bw_b0, WxT0, WhT0, bias0, EMBD);
  prep_weights_kernel<<<512, 256, 0, stream>>>(fw_W1, fw_b1, bw_W1, bw_b1, WxT1, WhT1, bias1, 512);

  dim3 g0(BB * TT / 128, G4 / 128, 2);
  // layer 0
  xw_gemm_kernel<EMBD><<<g0, 256, 0, stream>>>(x0, WxT0, bias0, XW);
  lstm_layer_kernel<<<2, 1024, SMEM_BYTES, stream>>>(XW, WhT0, lengths, y1);
  // layer 1 (input = concat(fw,bw) f16, K=512)
  xw_gemm_kernel<512><<<g0, 256, 0, stream>>>(y1, WxT1, bias1, XW);
  lstm_layer_kernel<<<2, 1024, SMEM_BYTES, stream>>>(XW, WhT1, lengths, y2);
  // output projection
  out_proj_kernel<<<(BB * TT * NOUT + 255) / 256, 256, 0, stream>>>(y2, out_W, out_b, (float*)d_out);
}
